// DecoderBlock_18356690223425
// MI455X (gfx1250) — compile-verified, matrix operand order checked by reading, not run on hardware
//
#include <hip/hip_runtime.h>

// ---------------------------------------------------------------------------
// Decoder block for MI455X (gfx1250): bf16 WMMA GEMMs + flash attention.
// B=8 S=512 E=768 H=12 HD=64 FF=3072.  wave32 only.
// Weights are pre-converted to bf16 in WMMA-fragment-swizzled tile order so
// GEMMs stage them with GLOBAL_LOAD_ASYNC_TO_LDS_B128 (ASYNCcnt) and read
// fragments with ds_load_b128 (no scalar u16 gathers / v_mov_b16 packing).
// ---------------------------------------------------------------------------

#define SEQ 512
#define EMB 768
#define NH  12
#define HDM 64
#define NB  8

typedef __attribute__((ext_vector_type(16))) __bf16 v16bf;
typedef __attribute__((ext_vector_type(8)))  float  v8f;

__device__ __forceinline__ float bf2f(__bf16 b) {
  union { unsigned short s; __bf16 b; } in; in.b = b;
  union { unsigned u; float f; } out; out.u = ((unsigned)in.s) << 16;
  return out.f;
}
__device__ __forceinline__ __bf16 f2bf(float f) {
  union { float f; unsigned u; } in; in.f = f;
  unsigned r = (in.u + 0x7FFFu + ((in.u >> 16) & 1u)) >> 16;   // RNE
  union { unsigned short s; __bf16 b; } out; out.s = (unsigned short)r;
  return out.b;
}
// pack two f32 -> dword of two bf16 (RNE), little-endian order {lo=a, hi=b}
__device__ __forceinline__ unsigned pack2(float a, float b) {
  union { float f; unsigned u; } x, y; x.f = a; y.f = b;
  unsigned lo = (x.u + 0x7FFFu + ((x.u >> 16) & 1u)) >> 16;
  unsigned hi = (y.u + 0x7FFFu + ((y.u >> 16) & 1u)) >> 16;
  return (lo & 0xFFFFu) | (hi << 16);
}
// A/B fragment K index for element j of v16bf (ISA 7.12.2, 16-bit 16x32)
__device__ __forceinline__ constexpr int kidx(int j, int half) {
  return ((j < 8) ? j : j + 8) + 8 * half;
}
// inverse: k32 in 0..31 -> (half, j)
__device__ __forceinline__ void kinv(int k32, int& half, int& j) {
  half = (k32 >> 3) & 1;
  const int t0 = k32 - 8 * half;           // in {0..7} u {16..23}
  j = (t0 < 8) ? t0 : t0 - 8;
}
__device__ __forceinline__ unsigned lds_off(const void* p) {
  return (unsigned)(size_t)p;              // low 32 bits of generic = LDS offset
}
__device__ __forceinline__ void async_copy_b128(unsigned lds_dst, const void* gsrc) {
  asm volatile("global_load_async_to_lds_b128 %0, %1, off"
               :: "v"(lds_dst), "v"((unsigned long long)(size_t)gsrc)
               : "memory");
}
__device__ __forceinline__ void wait_async0() {
  asm volatile("s_wait_asynccnt 0x0" ::: "memory");
}
__device__ __forceinline__ void wait_ds0() {
  asm volatile("s_wait_dscnt 0x0" ::: "memory");
}

// ---------------------------------------------------------------------------
// Weight pre-pass: W (K,N) f32 row-major -> bf16 tiles in fragment order.
// Tile (nt, kt) covers K rows [kt*32, kt*32+32) x N cols [nt*64, nt*64+64),
// stored contiguously (2048 bf16 = 4KB) at (nt*ktiles + kt), internal layout
// [csub(4)][lane(32)][j(16)] with value = W[kt*32 + kidx(j, lane>>4)]
//                                          [nt*64 + csub*16 + (lane&15)].
// One thread per dword (j, j+1).
// ---------------------------------------------------------------------------
__global__ __launch_bounds__(256)
void swizzle_weights(const float* __restrict__ W, __bf16* __restrict__ Wsw,
                     int K, int N) {
  const int idx  = blockIdx.x * 256 + threadIdx.x;   // dword index, K*N/2 total
  const int j    = (idx & 7) * 2;
  const int lane = (idx >> 3) & 31;
  const int c    = (idx >> 8) & 3;
  const int tile = idx >> 10;
  const int ktiles = K >> 5;
  const int nt = tile / ktiles, kt = tile - nt * ktiles;
  const int k = kt * 32 + kidx(j, lane >> 4);        // kidx(j+1)=k+1 (j even)
  const int n = nt * 64 + c * 16 + (lane & 15);
  ((unsigned*)Wsw)[idx] = pack2(W[(size_t)k * N + n], W[(size_t)(k + 1) * N + n]);
}

// ---------------------------------------------------------------------------
// GEMM: C(M,N) = A(M,K) @ W(K,N) + bias, optional ReLU. A f32 row-major,
// W pre-swizzled bf16 (above). 256 thr = 8 waves, 64x64 C tile, BK=32.
// B tile staged by async DMA; A tile converted+swizzled by the loader.
// Fragments = 32 contiguous LDS bytes per lane -> 2x ds_load_b128.
// ---------------------------------------------------------------------------
template<bool RELU>
__global__ __launch_bounds__(256)
void gemm_bf16_wmma(const float* __restrict__ A, const __bf16* __restrict__ Wsw,
                    const float* __restrict__ bias, float* __restrict__ C,
                    int M, int N, int K) {
  __shared__ __align__(32) __bf16 Asw[4][32][16];    // 4KB, fragment order
  __shared__ __align__(32) __bf16 Bsw[4][32][16];    // 4KB, async dest

  const int tid  = threadIdx.x;
  const int lane = tid & 31, half = lane >> 4, lr = lane & 15;
  const int wave = tid >> 5, wr = wave >> 1, wc = wave & 1;
  const int m0 = blockIdx.y * 64;
  const int ntile = blockIdx.x;
  const int ktiles = K >> 5;
  const unsigned bdst = lds_off(&Bsw[0][0][0]) + tid * 16;

  v8f acc0 = {0.f,0.f,0.f,0.f,0.f,0.f,0.f,0.f};
  v8f acc1 = acc0;

  for (int kt = 0; kt < ktiles; ++kt) {
    // async: 4KB pre-swizzled B tile, 16B per thread
    async_copy_b128(bdst, Wsw + (((size_t)ntile * ktiles + kt) << 11) + tid * 8);
    // A tile 64x32 f32 -> swizzled bf16 (1024 dword stores, 4 per thread)
#pragma unroll
    for (int i = 0; i < 4; ++i) {
      const int idx = tid + i * 256;
      const int r = idx >> 4, k = (idx & 15) * 2;
      const float2 av = *(const float2*)(A + (size_t)(m0 + r) * K + kt * 32 + k);
      int hk, j; kinv(k, hk, j);
      *(unsigned*)&Asw[r >> 4][(r & 15) + 16 * hk][j] = pack2(av.x, av.y);
    }
    wait_async0();
    __syncthreads();

    const v16bf a  = *(const v16bf*)&Asw[wr][lane][0];
    const v16bf b0 = *(const v16bf*)&Bsw[2 * wc + 0][lane][0];
    const v16bf b1 = *(const v16bf*)&Bsw[2 * wc + 1][lane][0];
    acc0 = __builtin_amdgcn_wmma_f32_16x16x32_bf16(false, a, false, b0, (short)0, acc0, false, false);
    acc1 = __builtin_amdgcn_wmma_f32_16x16x32_bf16(false, a, false, b1, (short)0, acc1, false, false);
    __syncthreads();
  }

#pragma unroll
  for (int j = 0; j < 8; ++j) {
    const int row = m0 + wr * 16 + j + 8 * half;
    const int c0 = ntile * 64 + (2 * wc + 0) * 16 + lr;
    const int c1 = ntile * 64 + (2 * wc + 1) * 16 + lr;
    float v0 = acc0[j] + bias[c0];
    float v1 = acc1[j] + bias[c1];
    if (RELU) { v0 = fmaxf(v0, 0.f); v1 = fmaxf(v1, 0.f); }
    C[(size_t)row * N + c0] = v0;
    C[(size_t)row * N + c1] = v1;
  }
}

// ---------------------------------------------------------------------------
// Flash attention per (b, h, 64-query block). 128 thr = 4 waves, each wave
// owns 16 query rows; streams 64-key blocks. Q/K/V/P kept in LDS in WMMA
// fragment order (b128 fragment loads). Optional XL relative-position bias:
// for logits[b,h,s',k'] with T=(h*512+s')*512+k', the reshaped-rpem dot is a
// contiguous 64-float dot with rpe_emb[min((T%6144)/12 - T/6144 + 512, 1022),
// 64*(T%12)..]; Q is also kept row-major in LDS for that dot.
// Out layout: (B,S,H,HD) flattened to (B*S, E).
// ---------------------------------------------------------------------------
template<bool WITH_RPE>
__global__ __launch_bounds__(128)
void flash_attn(const float* __restrict__ Qb, int q_rs, int q_hs,
                const float* __restrict__ Kb, int k_rs, int k_hs,
                const float* __restrict__ Vb, int v_rs, int v_hs,
                const float* __restrict__ rpe,
                float* __restrict__ Out, float scale) {
  __shared__ __align__(32) __bf16 Qsw[4][2][32][16];   // [wave][frag]
  __shared__ __align__(32) __bf16 Ksw[4][2][32][16];   // [keysub][frag]
  __shared__ __align__(32) __bf16 Vsw[4][2][32][16];   // [dsub][frag]
  __shared__ __align__(32) __bf16 Psw[4][2][32][16];   // [wave][frag]
  __shared__ __bf16 Qrm[64][68];                       // row-major for rpe dot

  const int tid  = threadIdx.x;
  const int lane = tid & 31, half = lane >> 4, lr = lane & 15;
  const int wave = tid >> 5;
  const int bid = blockIdx.x;
  const int qb = bid & 7;                 // S/64 = 8 query blocks
  const int h  = (bid >> 3) % NH;
  const int b  = bid / (8 * NH);

  const float* Qp = Qb + (size_t)b * SEQ * q_rs + h * q_hs;
  const float* Kp = Kb + (size_t)b * SEQ * k_rs + h * k_hs;
  const float* Vp = Vb + (size_t)b * SEQ * v_rs + h * v_hs;

  // ---- stage Q (swizzled + optional row-major copy) ----
#pragma unroll
  for (int i = 0; i < 16; ++i) {
    const int idx = tid + i * 128;        // 2048 dwords = 64 rows x 64 d
    const int r = idx >> 5, k = (idx & 31) * 2;
    const float2 qv = *(const float2*)(Qp + (size_t)(qb * 64 + r) * q_rs + k);
    const unsigned d = pack2(qv.x, qv.y);
    int hk, j; kinv(k & 31, hk, j);
    *(unsigned*)&Qsw[r >> 4][k >> 5][(r & 15) + 16 * hk][j] = d;
    if (WITH_RPE) *(unsigned*)&Qrm[r][k] = d;
  }
  __syncthreads();

  const v16bf qf0 = *(const v16bf*)&Qsw[wave][0][lane][0];
  const v16bf qf1 = *(const v16bf*)&Qsw[wave][1][lane][0];

  float mrow[8], lrow[8];
#pragma unroll
  for (int j = 0; j < 8; ++j) { mrow[j] = -3.0e38f; lrow[j] = 0.f; }
  const v8f z = {0.f,0.f,0.f,0.f,0.f,0.f,0.f,0.f};
  v8f o[4] = {z, z, z, z};                // 16 x 64 output accumulator

  for (int t = 0; t < 8; ++t) {           // 8 key blocks of 64
    // ---- stage K (row-swizzle, like Q) and V (B-type over keys) ----
#pragma unroll
    for (int i = 0; i < 16; ++i) {
      const int idx = tid + i * 128;
      const int r = idx >> 5, k = (idx & 31) * 2;
      const float2 kv = *(const float2*)(Kp + (size_t)(t * 64 + r) * k_rs + k);
      int hk, j; kinv(k & 31, hk, j);
      *(unsigned*)&Ksw[r >> 4][k >> 5][(r & 15) + 16 * hk][j] = pack2(kv.x, kv.y);
    }
#pragma unroll
    for (int i = 0; i < 16; ++i) {
      const int idx = tid + i * 128;
      const int d = idx >> 5, key = (idx & 31) * 2;
      const float v0 = Vp[(size_t)(t * 64 + key) * v_rs + d];
      const float v1 = Vp[(size_t)(t * 64 + key + 1) * v_rs + d];
      int hk, j; kinv(key & 31, hk, j);
      *(unsigned*)&Vsw[d >> 4][key >> 5][(d & 15) + 16 * hk][j] = pack2(v0, v1);
    }
    __syncthreads();

    // ---- S = Q @ K^T (16x64 per wave), + scale, + rpe bias ----
    v8f st[4];
#pragma unroll
    for (int n = 0; n < 4; ++n) {
      v8f s = z;
      const v16bf kb0 = *(const v16bf*)&Ksw[n][0][lane][0];
      const v16bf kb1 = *(const v16bf*)&Ksw[n][1][lane][0];
      s = __builtin_amdgcn_wmma_f32_16x16x32_bf16(false, qf0, false, kb0, (short)0, s, false, false);
      s = __builtin_amdgcn_wmma_f32_16x16x32_bf16(false, qf1, false, kb1, (short)0, s, false, false);
#pragma unroll
      for (int j = 0; j < 8; ++j) {
        float val = s[j] * scale;
        if (WITH_RPE) {
          const int sq = qb * 64 + wave * 16 + j + 8 * half;   // query row
          const int kc = t * 64 + n * 16 + lr;                 // key col
          const int T  = (h * 512 + sq) * 512 + kc;
          const int ssrc = T / 6144;
          const int krow = (T % 6144) / 12;
          const int e0   = (T % 12) * 64;
          const int ridx = min(krow - ssrc + 512, 1022);       // mode='clip'
          const float* rp = rpe + (size_t)ridx * EMB + e0;
          __builtin_prefetch(rp, 0, 0);                        // global_prefetch_b8
          const int ql = wave * 16 + j + 8 * half;
          float racc = 0.f;
#pragma unroll 4
          for (int d = 0; d < 64; ++d)
            racc += bf2f(Qrm[ql][d]) * rp[d];
          val += racc;
        }
        st[n][j] = val;
      }
    }

    // ---- online softmax over the 64 new keys ----
    float rmax[8], alpha[8], rsum[8];
#pragma unroll
    for (int j = 0; j < 8; ++j) {
      float m = fmaxf(fmaxf(st[0][j], st[1][j]), fmaxf(st[2][j], st[3][j]));
#pragma unroll
      for (int msk = 1; msk <= 8; msk <<= 1)   // within 16-lane row group
        m = fmaxf(m, __shfl_xor(m, msk, 32));
      rmax[j] = m;
    }
#pragma unroll
    for (int j = 0; j < 8; ++j) {
      const float mnew = fmaxf(mrow[j], rmax[j]);
      alpha[j] = __expf(mrow[j] - mnew);
      mrow[j] = mnew;
      rsum[j] = 0.f;
    }
#pragma unroll
    for (int n = 0; n < 4; ++n)
#pragma unroll
      for (int j = 0; j < 8; ++j) {
        const float p = __expf(st[n][j] - mrow[j]);
        st[n][j] = p;
        rsum[j] += p;
      }
#pragma unroll
    for (int j = 0; j < 8; ++j) {
      float ssm = rsum[j];
#pragma unroll
      for (int msk = 1; msk <= 8; msk <<= 1)
        ssm += __shfl_xor(ssm, msk, 32);
      lrow[j] = lrow[j] * alpha[j] + ssm;
    }
#pragma unroll
    for (int u = 0; u < 4; ++u)
#pragma unroll
      for (int j = 0; j < 8; ++j) o[u][j] *= alpha[j];

    // ---- P: C layout -> A-fragment layout via per-wave LDS scratch ----
#pragma unroll
    for (int n = 0; n < 4; ++n)
#pragma unroll
      for (int j2 = 0; j2 < 8; ++j2) {
        const int m   = j2 + 8 * half;           // P row
        const int k32 = (n & 1) * 16 + lr;       // key within 32-chunk
        int hd, jd; kinv(k32, hd, jd);
        Psw[wave][n >> 1][m + 16 * hd][jd] = f2bf(st[n][j2]);
      }
    wait_ds0();                                  // wave-local LDS RAW
    const v16bf p0 = *(const v16bf*)&Psw[wave][0][lane][0];
    const v16bf p1 = *(const v16bf*)&Psw[wave][1][lane][0];

    // ---- O += P @ V ----
#pragma unroll
    for (int u = 0; u < 4; ++u) {
      const v16bf vb0 = *(const v16bf*)&Vsw[u][0][lane][0];
      const v16bf vb1 = *(const v16bf*)&Vsw[u][1][lane][0];
      o[u] = __builtin_amdgcn_wmma_f32_16x16x32_bf16(false, p0, false, vb0, (short)0, o[u], false, false);
      o[u] = __builtin_amdgcn_wmma_f32_16x16x32_bf16(false, p1, false, vb1, (short)0, o[u], false, false);
    }
    __syncthreads();   // before next iteration restages Ksw/Vsw
  }

#pragma unroll
  for (int u = 0; u < 4; ++u)
#pragma unroll
    for (int j = 0; j < 8; ++j) {
      const int sq = qb * 64 + wave * 16 + j + 8 * half;
      const int d  = u * 16 + lr;
      Out[(size_t)(b * SEQ + sq) * EMB + h * HDM + d] = o[u][j] / lrow[j];
    }
}

// ---------------------------------------------------------------------------
// out = LayerNorm(a + b) * gamma + beta over rows of length 768.
// ---------------------------------------------------------------------------
__global__ __launch_bounds__(256)
void add_ln(const float* __restrict__ a, const float* __restrict__ r,
            const float* __restrict__ g, const float* __restrict__ be,
            float* __restrict__ out) {
  __shared__ float ssum[256], ssq[256];
  const int row = blockIdx.x, tid = threadIdx.x;
  const size_t base = (size_t)row * EMB;
  const float x0 = a[base + tid]       + r[base + tid];
  const float x1 = a[base + tid + 256] + r[base + tid + 256];
  const float x2 = a[base + tid + 512] + r[base + tid + 512];
  ssum[tid] = x0 + x1 + x2;
  ssq[tid]  = x0 * x0 + x1 * x1 + x2 * x2;
  __syncthreads();
  for (int off = 128; off > 0; off >>= 1) {
    if (tid < off) { ssum[tid] += ssum[tid + off]; ssq[tid] += ssq[tid + off]; }
    __syncthreads();
  }
  const float mean = ssum[0] * (1.0f / 768.0f);
  const float var  = ssq[0] * (1.0f / 768.0f) - mean * mean;
  const float inv  = rsqrtf(var + 1e-6f);
  out[base + tid]       = (x0 - mean) * inv * g[tid]       + be[tid];
  out[base + tid + 256] = (x1 - mean) * inv * g[tid + 256] + be[tid + 256];
  out[base + tid + 512] = (x2 - mean) * inv * g[tid + 512] + be[tid + 512];
}

// ---------------------------------------------------------------------------
extern "C" void kernel_launch(void* const* d_in, const int* in_sizes, int n_in,
                              void* d_out, int out_size, void* d_ws, size_t ws_size,
                              hipStream_t stream) {
  (void)in_sizes; (void)n_in; (void)out_size; (void)ws_size;
  const float* x     = (const float*)d_in[0];
  const float* enc   = (const float*)d_in[1];
  const float* qkv_w = (const float*)d_in[2];
  const float* qkv_b = (const float*)d_in[3];
  const float* so_w  = (const float*)d_in[4];
  const float* so_b  = (const float*)d_in[5];
  const float* rpe   = (const float*)d_in[6];
  const float* cq_w  = (const float*)d_in[7];
  const float* cq_b  = (const float*)d_in[8];
  const float* ckv_w = (const float*)d_in[9];
  const float* ckv_b = (const float*)d_in[10];
  const float* co_w  = (const float*)d_in[11];
  const float* co_b  = (const float*)d_in[12];
  const float* ff1_w = (const float*)d_in[13];
  const float* ff1_b = (const float*)d_in[14];
  const float* ff2_w = (const float*)d_in[15];
  const float* ff2_b = (const float*)d_in[16];
  const float* ln1_s = (const float*)d_in[17];
  const float* ln1_b = (const float*)d_in[18];
  const float* ln2_s = (const float*)d_in[19];
  const float* ln2_b = (const float*)d_in[20];
  const float* ln3_s = (const float*)d_in[21];
  const float* ln3_b = (const float*)d_in[22];
  float* outp = (float*)d_out;

  const int M = NB * SEQ;                 // 4096 rows
  float* w    = (float*)d_ws;             // f32 scratch + bf16 weights (~157 MB)
  float* big  = w;                        // 12,582,912: qkv (9.44M) then ffh
  float* bufA = big  + 12582912;          //  3,145,728: attn vals (self/cross)
  float* bufB = bufA + 3145728;           //  3,145,728: sa_out / ca / ff2_out
  float* h1   = bufB + 3145728;
  float* h2   = h1   + 3145728;
  float* cqb  = h2   + 3145728;
  float* ckvb = cqb  + 3145728;           //  6,291,456
  __bf16* wsw  = (__bf16*)(ckvb + 6291456);
  __bf16* qkvW = wsw;                     // 1,769,472
  __bf16* soW  = qkvW + 1769472;          //   589,824
  __bf16* cqW  = soW  + 589824;           //   589,824
  __bf16* ckvW = cqW  + 589824;           // 1,179,648
  __bf16* coW  = ckvW + 1179648;          //   589,824
  __bf16* ff1W = coW  + 589824;           // 2,359,296
  __bf16* ff2W = ff1W + 2359296;          // 2,359,296

  const float scale = 0.125f;             // 1/sqrt(64)
  const dim3 blk(256);
  const dim3 ablk(128);
  const dim3 agrid(8 * NH * NB);          // 768 blocks

  // ---- weight pre-pass: f32 -> bf16 fragment-swizzled tiles ----
  swizzle_weights<<<dim3(768*2304/512), blk, 0, stream>>>(qkv_w, qkvW, 768, 2304);
  swizzle_weights<<<dim3(768*768 /512), blk, 0, stream>>>(so_w,  soW,  768, 768);
  swizzle_weights<<<dim3(768*768 /512), blk, 0, stream>>>(cq_w,  cqW,  768, 768);
  swizzle_weights<<<dim3(768*1536/512), blk, 0, stream>>>(ckv_w, ckvW, 768, 1536);
  swizzle_weights<<<dim3(768*768 /512), blk, 0, stream>>>(co_w,  coW,  768, 768);
  swizzle_weights<<<dim3(768*3072/512), blk, 0, stream>>>(ff1_w, ff1W, 768, 3072);
  swizzle_weights<<<dim3(3072*768/512), blk, 0, stream>>>(ff2_w, ff2W, 3072, 768);

  // ---- self-attention ----
  gemm_bf16_wmma<false><<<dim3(2304/64, M/64), blk, 0, stream>>>(x, qkvW, qkv_b, big, M, 2304, 768);
  flash_attn<true><<<agrid, ablk, 0, stream>>>(big, 2304, 192, big + 64, 2304, 192,
                                               big + 128, 2304, 192, rpe, bufA, scale);
  gemm_bf16_wmma<false><<<dim3(768/64, M/64), blk, 0, stream>>>(bufA, soW, so_b, bufB, M, 768, 768);
  add_ln<<<dim3(M), blk, 0, stream>>>(x, bufB, ln1_s, ln1_b, h1);

  // ---- cross-attention ----
  gemm_bf16_wmma<false><<<dim3(768/64,  M/64), blk, 0, stream>>>(h1,  cqW,  cq_b,  cqb,  M, 768,  768);
  gemm_bf16_wmma<false><<<dim3(1536/64, M/64), blk, 0, stream>>>(enc, ckvW, ckv_b, ckvb, M, 1536, 768);
  flash_attn<false><<<agrid, ablk, 0, stream>>>(cqb, 768, 64, ckvb, 1536, 128,
                                                ckvb + 64, 1536, 128, nullptr, bufA, scale);
  gemm_bf16_wmma<false><<<dim3(768/64, M/64), blk, 0, stream>>>(bufA, coW, co_b, bufB, M, 768, 768);
  add_ln<<<dim3(M), blk, 0, stream>>>(h1, bufB, ln2_s, ln2_b, h2);

  // ---- feed-forward ----
  gemm_bf16_wmma<true ><<<dim3(3072/64, M/64), blk, 0, stream>>>(h2,  ff1W, ff1_b, big,  M, 3072, 768);
  gemm_bf16_wmma<false><<<dim3(768/64,  M/64), blk, 0, stream>>>(big, ff2W, ff2_b, bufB, M, 768, 3072);
  add_ln<<<dim3(M), blk, 0, stream>>>(h2, bufB, ln3_s, ln3_b, outp);
}
